// Lowrank_Spattention_47021301956920
// MI455X (gfx1250) — matrix operand order repaired and checked by reading, not compiled, not run on hardware
//
#include <hip/hip_runtime.h>
#include <hip/hip_bf16.h>

typedef __attribute__((ext_vector_type(16))) _Float16 v16h;
typedef __attribute__((ext_vector_type(8)))  _Float16 v8h;
typedef __attribute__((ext_vector_type(4)))  _Float16 v4h;
typedef __attribute__((ext_vector_type(8)))  float    v8f;
typedef __attribute__((ext_vector_type(4)))  unsigned int u32x4;
typedef __attribute__((ext_vector_type(8)))  int      i32x8;
typedef __attribute__((ext_vector_type(4)))  int      i32x4;

#define HEADS 4
#define RANK 32
#define HD 32          // head_dim
#define FD 128         // feature dim
#define CH 128         // rows per chunk
#define NCHUNK 32      // 4096 / 128
#define NSLAB 96       // B*T
#define SCALE 0.17677669529663687f  // 1/sqrt(32)

#if defined(__has_builtin)
#if __has_builtin(__builtin_amdgcn_tensor_load_to_lds) && \
    __has_builtin(__builtin_amdgcn_tensor_store_from_lds) && \
    __has_builtin(__builtin_amdgcn_s_wait_tensorcnt)
#define USE_TDM 1
#endif
#endif
#ifndef USE_TDM
#define USE_TDM 0
#endif

union V16U { v16h v; v8h h[2]; };

// A-operand (MxK, K=32 tile) from row-major [M][ld] f16, base at (m0,k0).
// lane l: M = l%16 ; elems 0..7 -> K = (l/16)*8 + 0..7 ; elems 8..15 -> K += 16
__device__ inline v16h afrag(const _Float16* base, int ld, int lane) {
  const _Float16* p = base + (lane & 15) * ld + ((lane >> 4) << 3);
  V16U u;
  u.h[0] = *(const v8h*)p;
  u.h[1] = *(const v8h*)(p + 16);
  return u.v;
}

// Same A-operand pattern, but source is row-major f32 (convert at build time).
__device__ inline v16h afrag_cvt(const float* base, int ld, int lane) {
  const float* p = base + (lane & 15) * ld + ((lane >> 4) << 3);
  v16h r;
#pragma unroll
  for (int e = 0; e < 8; e++) r[e] = (_Float16)p[e];
#pragma unroll
  for (int e = 0; e < 8; e++) r[e + 8] = (_Float16)p[e + 16];
  return r;
}

// B-operand (KxN, K=32 tile) from row-major B^T [N][ld] f16, base at (n0,k0).
// lane l: N = l%16 ; elems 0..7 -> K = (l/16)*8 + 0..7 ; elems 8..15 -> K += 16
__device__ inline v16h bfrag(const _Float16* base, int ld, int lane) {
  const _Float16* p = base + (lane & 15) * ld + ((lane >> 4) << 3);
  V16U u;
  u.h[0] = *(const v8h*)p;
  u.h[1] = *(const v8h*)(p + 16);
  return u.v;
}

__device__ inline v8f wmma16(v16h a, v16h b, v8f c) {
  return __builtin_amdgcn_wmma_f32_16x16x32_f16(false, a, false, b, (short)0, c,
                                                false, false);
}

#if USE_TDM
// Build a 2-D f32-tile TDM descriptor (D#) and issue the DMA.
// Group0: count=1 | lds_addr[63:32] | global_addr[120:64] | type=2[127:126]
// Group1: data_size=4B[17:16] | tensor_dim0[79:48] | tensor_dim1[111:80]
//         | tile_dim0[127:112] | tile_dim1[143:128] | dim0_stride[207:160]
__device__ inline void tdm_desc(u32x4& g0, i32x8& g1, unsigned lds_off,
                                const void* gaddr, int rows, int cols) {
  unsigned long long ga = (unsigned long long)gaddr;
  g0[0] = 1u;                                             // count = 1 (valid)
  g0[1] = lds_off;                                        // lds_addr
  g0[2] = (unsigned)(ga & 0xffffffffu);                   // global_addr[31:0]
  g0[3] = (unsigned)((ga >> 32) & 0x1ffffffu) | (2u << 30); // addr[56:32] | type=2
  g1[0] = (int)(2u << 16);          // data_size = 2 (4 bytes)
  g1[1] = (int)((unsigned)cols << 16);   // tensor_dim0 lo16 @ bits[63:48]
  g1[2] = (int)((unsigned)rows << 16);   // tensor_dim1 lo16 @ bits[95:80]
  g1[3] = (int)((unsigned)cols << 16);   // tile_dim0        @ bits[127:112]
  g1[4] = rows;                          // tile_dim1        @ bits[143:128]
  g1[5] = cols;                          // tensor_dim0_stride lo32
  g1[6] = 0;
  g1[7] = 0;
}

__device__ inline void tdm_load_tile(unsigned lds_off, const void* gaddr,
                                     int rows, int cols) {
  u32x4 g0; i32x8 g1;
  tdm_desc(g0, g1, lds_off, gaddr, rows, cols);
  i32x4 z4 = {0, 0, 0, 0};
#if __clang_major__ >= 23
  i32x8 z8 = {0, 0, 0, 0, 0, 0, 0, 0};
  __builtin_amdgcn_tensor_load_to_lds(g0, g1, z4, z4, z8, 0);
#else
  __builtin_amdgcn_tensor_load_to_lds(g0, g1, z4, z4, 0);
#endif
}

__device__ inline void tdm_store_tile(unsigned lds_off, void* gaddr,
                                      int rows, int cols) {
  u32x4 g0; i32x8 g1;
  tdm_desc(g0, g1, lds_off, gaddr, rows, cols);
  i32x4 z4 = {0, 0, 0, 0};
#if __clang_major__ >= 23
  i32x8 z8 = {0, 0, 0, 0, 0, 0, 0, 0};
  __builtin_amdgcn_tensor_store_from_lds(g0, g1, z4, z4, z8, 0);
#else
  __builtin_amdgcn_tensor_store_from_lds(g0, g1, z4, z4, 0);
#endif
}
#endif  // USE_TDM

// ---------------- k0: convert weights to f16 ----------------
__global__ void k0_prep(const float* __restrict__ Wq, const float* __restrict__ Wv,
                        const float* __restrict__ key_param,
                        _Float16* wqh, _Float16* wvh, _Float16* keyh) {
  int tid = blockIdx.x * blockDim.x + threadIdx.x;
  int stride = gridDim.x * blockDim.x;
  for (int i = tid; i < FD * FD; i += stride) {
    wqh[i] = (_Float16)Wq[i];
    wvh[i] = (_Float16)Wv[i];
  }
  // key_param [r][h][d] -> keyh [h][r][d]   (B^T layout for attn GEMM)
  for (int i = tid; i < HEADS * RANK * HD; i += stride) {
    int h = i >> 10, r = (i >> 5) & 31, d = i & 31;
    keyh[h * 1024 + r * 32 + d] = (_Float16)key_param[r * 128 + h * 32 + d];
  }
}

// ---------------- k1: q, attn, p1, xv, partial v1, partial col stats ----------------
__global__ void __launch_bounds__(256)
k1_stats(const float* __restrict__ x, const _Float16* __restrict__ wqh,
         const _Float16* __restrict__ wvh, const _Float16* __restrict__ keyh,
         const float* __restrict__ bq, const float* __restrict__ bv,
         float* __restrict__ part_v1, float* __restrict__ part_ms) {
  __shared__ float    xs[CH * FD];      // 64 KB   x chunk f32 (TDM target)
  __shared__ _Float16 qh[CH * FD];      // 32 KB   q chunk, f16
  __shared__ float    at[CH * FD];      // 64 KB   attn f32 [n][h*32+r]
  __shared__ _Float16 p1T[HEADS * RANK * CH]; // 32 KB  p1 transposed [h][r][n]
  __shared__ _Float16 xvT[FD * CH];     // 32 KB   xv transposed [f][n]

  const int w = threadIdx.x >> 5, lane = threadIdx.x & 31;
  const int slab = blockIdx.y, chunk = blockIdx.x;
  const int bl = slab * NCHUNK + chunk;

  // Phase A: x chunk (contiguous 128x128 f32) -> LDS
  const float* xp = x + ((size_t)slab * 4096 + (size_t)chunk * CH) * FD;
#if USE_TDM
  if (threadIdx.x < 32) {   // one wave issues the DMA; TDM ignores EXEC
    tdm_load_tile((unsigned)(size_t)(void*)xs, xp, CH, FD);
    __builtin_amdgcn_s_wait_tensorcnt(0);
  }
#else
  for (int i = threadIdx.x; i < CH * FD / 4; i += 256) {
    float4 f = ((const float4*)xp)[i];
    ((float4*)xs)[i] = f;
  }
#endif
  __syncthreads();

  // A fragments of x for this wave's 16-row m-tile (f32 -> f16 at build time)
  v16h afr[4];
#pragma unroll
  for (int kt = 0; kt < 4; kt++)
    afr[kt] = afrag_cvt(&xs[(w * 16) * FD + kt * 32], FD, lane);

  // Phase B: q = x @ Wq^T + bq  -> qh (f16)
#pragma unroll
  for (int j = 0; j < 8; j++) {
    v8f acc = {};
#pragma unroll
    for (int kt = 0; kt < 4; kt++) {
      v16h b = bfrag(wqh + (j * 16) * FD + kt * 32, FD, lane);
      acc = wmma16(afr[kt], b, acc);
    }
    int col = j * 16 + (lane & 15);
    float bias = bq[col];
    int rb = w * 16 + ((lane >> 4) << 3);
#pragma unroll
    for (int v = 0; v < 8; v++) qh[(rb + v) * FD + col] = (_Float16)(acc[v] + bias);
  }
  __syncthreads();

  // Phase C: attn[n][h*32+r] = (q_h . key_h^T) * scale
#pragma unroll
  for (int h = 0; h < HEADS; h++) {
    v16h aq = afrag(&qh[(w * 16) * FD + h * 32], FD, lane);
#pragma unroll
    for (int rt = 0; rt < 2; rt++) {
      v8f acc = {};
      v16h b = bfrag(keyh + (h * 32 + rt * 16) * 32, 32, lane);
      acc = wmma16(aq, b, acc);
      int c = h * 32 + rt * 16 + (lane & 15);
      int rb = w * 16 + ((lane >> 4) << 3);
#pragma unroll
      for (int v = 0; v < 8; v++) at[(rb + v) * FD + c] = acc[v] * SCALE;
    }
  }
  __syncthreads();

  // Phase D1: row softmax over r (32) per (row, head) -> p1T [h][r][n] f16
  for (int p = threadIdx.x; p < CH * HEADS; p += 256) {
    int row = p >> 2, h = p & 3;
    const float* a = &at[row * FD + h * 32];
    float mx = -1e30f;
    for (int r = 0; r < RANK; r++) mx = fmaxf(mx, a[r]);
    float s = 0.f;
    for (int r = 0; r < RANK; r++) s += __expf(a[r] - mx);
    float inv = 1.f / s;
    for (int r = 0; r < RANK; r++)
      p1T[(h * 32 + r) * CH + row] = (_Float16)(__expf(a[r] - mx) * inv);
  }
  // Phase D2: per-chunk column stats (max / sum-exp over the 128 rows)
  if (threadIdx.x < FD) {
    int c = threadIdx.x;
    float mx = -1e30f;
    for (int n = 0; n < CH; n++) mx = fmaxf(mx, at[n * FD + c]);
    float s = 0.f;
    for (int n = 0; n < CH; n++) s += __expf(at[n * FD + c] - mx);
    part_ms[(size_t)bl * 256 + c] = mx;
    part_ms[(size_t)bl * 256 + 128 + c] = s;
  }

  // Phase E: xv = x @ Wv^T + bv -> xvT [f][n] f16 (B^T layout for stage-1)
#pragma unroll
  for (int j = 0; j < 8; j++) {
    v8f acc = {};
#pragma unroll
    for (int kt = 0; kt < 4; kt++) {
      v16h b = bfrag(wvh + (j * 16) * FD + kt * 32, FD, lane);
      acc = wmma16(afr[kt], b, acc);
    }
    int f = j * 16 + (lane & 15);
    float bias = bv[f];
    v8h pk;
#pragma unroll
    for (int v = 0; v < 8; v++) pk[v] = (_Float16)(acc[v] + bias);
    *(v8h*)&xvT[f * CH + w * 16 + ((lane >> 4) << 3)] = pk;
  }
  __syncthreads();

  // Phase F: partial v1[r,h,d] = sum_n p1[n,h,r]*xv[n,h,d]  (per chunk)
#pragma unroll
  for (int ti = 0; ti < 2; ti++) {
    int t = w + ti * 8;                      // 16 tile-tasks over 8 waves
    int h = t >> 2, rt = (t >> 1) & 1, dt = t & 1;
    v8f acc = {};
#pragma unroll
    for (int kt = 0; kt < 4; kt++) {
      v16h a = afrag(&p1T[(h * 32 + rt * 16) * CH + kt * 32], CH, lane);
      v16h b = bfrag(&xvT[(h * 32 + dt * 16) * CH + kt * 32], CH, lane);
      acc = wmma16(a, b, acc);
    }
    float* pv = part_v1 + (size_t)bl * 4096 + h * 1024 + (rt * 16) * 32 + dt * 16;
    int r = (lane >> 4) << 3, d = lane & 15;
#pragma unroll
    for (int v = 0; v < 8; v++) pv[(r + v) * 32 + d] = acc[v];
  }
}

// ---------------- k2: deterministic reductions over chunks ----------------
__global__ void __launch_bounds__(256)
k2_reduce(const float* __restrict__ part_v1, const float* __restrict__ part_ms,
          float* __restrict__ msf, _Float16* __restrict__ v1T) {
  int slab = blockIdx.x;
  if (threadIdx.x < FD) {  // merge (max, sumexp) with log-sum-exp combine
    int c = threadIdx.x;
    float m = -1e30f, s = 0.f;
    for (int ch = 0; ch < NCHUNK; ch++) {
      float mc = part_ms[((size_t)slab * NCHUNK + ch) * 256 + c];
      float sc = part_ms[((size_t)slab * NCHUNK + ch) * 256 + 128 + c];
      if (mc > m) { s = s * __expf(m - mc) + sc; m = mc; }
      else        { s += sc * __expf(mc - m); }
    }
    msf[(size_t)slab * 256 + c] = m;
    msf[(size_t)slab * 256 + 128 + c] = s;
  }
  for (int i = threadIdx.x; i < 4096; i += 256) {
    float acc = 0.f;
    for (int ch = 0; ch < NCHUNK; ch++)
      acc += part_v1[((size_t)slab * NCHUNK + ch) * 4096 + i];
    int h = i >> 10, r = (i >> 5) & 31, d = i & 31;
    // store transposed [h][d][r] f16: B^T layout for stage-2 GEMM
    v1T[(size_t)slab * 4096 + h * 1024 + d * 32 + r] = (_Float16)acc;
  }
}

// ---------------- k3: recompute q/attn, p2, stage-2, gate, output ----------------
__global__ void __launch_bounds__(256)
k3_out(const float* __restrict__ x, const _Float16* __restrict__ wqh,
       const _Float16* __restrict__ wvh, const _Float16* __restrict__ keyh,
       const float* __restrict__ bq, const float* __restrict__ bv,
       const float* __restrict__ alpha, const float* __restrict__ beta,
       const float* __restrict__ msf, const _Float16* __restrict__ v1T,
       float* __restrict__ out) {
  __shared__ float    xs[CH * FD];    // 64 KB  x chunk f32 (TDM target)
  __shared__ _Float16 qh[CH * FD];    // 32 KB
  __shared__ _Float16 p2s[CH * FD];   // 32 KB  p2 f16 [n][h*32+r]
  __shared__ float    ob[CH * FD];    // 64 KB  output staging (TDM source)

  const int w = threadIdx.x >> 5, lane = threadIdx.x & 31;
  const int slab = blockIdx.y, chunk = blockIdx.x;

  const float* xp = x + ((size_t)slab * 4096 + (size_t)chunk * CH) * FD;
#if USE_TDM
  if (threadIdx.x < 32) {
    tdm_load_tile((unsigned)(size_t)(void*)xs, xp, CH, FD);
    __builtin_amdgcn_s_wait_tensorcnt(0);
  }
#else
  for (int i = threadIdx.x; i < CH * FD / 4; i += 256)
    ((float4*)xs)[i] = ((const float4*)xp)[i];
#endif
  __syncthreads();

  v16h afr[4];
#pragma unroll
  for (int kt = 0; kt < 4; kt++)
    afr[kt] = afrag_cvt(&xs[(w * 16) * FD + kt * 32], FD, lane);

  // q GEMM -> qh
#pragma unroll
  for (int j = 0; j < 8; j++) {
    v8f acc = {};
#pragma unroll
    for (int kt = 0; kt < 4; kt++) {
      v16h b = bfrag(wqh + (j * 16) * FD + kt * 32, FD, lane);
      acc = wmma16(afr[kt], b, acc);
    }
    int col = j * 16 + (lane & 15);
    float bias = bq[col];
    int rb = w * 16 + ((lane >> 4) << 3);
#pragma unroll
    for (int v = 0; v < 8; v++) qh[(rb + v) * FD + col] = (_Float16)(acc[v] + bias);
  }
  __syncthreads();

  // attn GEMM; convert directly to p2 = exp(attn - m_col)/s_col (column softmax)
#pragma unroll
  for (int h = 0; h < HEADS; h++) {
    v16h aq = afrag(&qh[(w * 16) * FD + h * 32], FD, lane);
#pragma unroll
    for (int rt = 0; rt < 2; rt++) {
      v8f acc = {};
      v16h b = bfrag(keyh + (h * 32 + rt * 16) * 32, 32, lane);
      acc = wmma16(aq, b, acc);
      int c = h * 32 + rt * 16 + (lane & 15);
      float mC = msf[(size_t)slab * 256 + c];
      float inv = 1.f / msf[(size_t)slab * 256 + 128 + c];
      int rb = w * 16 + ((lane >> 4) << 3);
#pragma unroll
      for (int v = 0; v < 8; v++)
        p2s[(rb + v) * FD + c] = (_Float16)(__expf(acc[v] * SCALE - mC) * inv);
    }
  }
  __syncthreads();

  float sa[HEADS], sb[HEADS];
#pragma unroll
  for (int h = 0; h < HEADS; h++) {
    sa[h] = 1.f / (1.f + __expf(-alpha[h]));
    sb[h] = 1.f / (1.f + __expf(-beta[h]));
  }

  // per feature tile: v2 (K=32 WMMA) + xv (K=128 WMMA) + gated mix -> ob
#pragma unroll
  for (int j = 0; j < 8; j++) {
    int h = j >> 1, dt = j & 1;
    v16h ap = afrag(&p2s[(w * 16) * FD + h * 32], FD, lane);
    v16h bv1 = bfrag(v1T + (size_t)slab * 4096 + h * 1024 + (dt * 16) * 32, 32, lane);
    v8f v2a = {};
    v2a = wmma16(ap, bv1, v2a);
    v8f xva = {};
#pragma unroll
    for (int kt = 0; kt < 4; kt++) {
      v16h b = bfrag(wvh + (j * 16) * FD + kt * 32, FD, lane);
      xva = wmma16(afr[kt], b, xva);
    }
    int col = j * 16 + (lane & 15);
    float bias = bv[col];
    int rb = w * 16 + ((lane >> 4) << 3);
#pragma unroll
    for (int v = 0; v < 8; v++)
      ob[(rb + v) * FD + col] = sa[h] * (xva[v] + bias) + sb[h] * v2a[v];
  }
  __syncthreads();

  float* op = out + ((size_t)slab * 4096 + (size_t)chunk * CH) * FD;
#if USE_TDM
  if (threadIdx.x < 32) {
    tdm_store_tile((unsigned)(size_t)(void*)ob, op, CH, FD);
    __builtin_amdgcn_s_wait_tensorcnt(0);  // S_ENDPGM also implies wait-idle
  }
#else
  for (int i = threadIdx.x; i < CH * FD / 4; i += 256)
    ((float4*)op)[i] = ((const float4*)ob)[i];
#endif
}

extern "C" void kernel_launch(void* const* d_in, const int* in_sizes, int n_in,
                              void* d_out, int out_size, void* d_ws, size_t ws_size,
                              hipStream_t stream) {
  const float* x    = (const float*)d_in[0];
  const float* Wq   = (const float*)d_in[1];
  const float* bq   = (const float*)d_in[2];
  const float* keyp = (const float*)d_in[3];
  const float* Wv   = (const float*)d_in[4];
  const float* bv   = (const float*)d_in[5];
  const float* alpha= (const float*)d_in[6];
  const float* beta = (const float*)d_in[7];
  float* out = (float*)d_out;

  char* ws = (char*)d_ws;
  _Float16* wqh   = (_Float16*)(ws + 0);          //  32 KB
  _Float16* wvh   = (_Float16*)(ws + 32768);      //  32 KB
  _Float16* keyh  = (_Float16*)(ws + 65536);      //   8 KB
  _Float16* v1T   = (_Float16*)(ws + 73728);      // 768 KB
  float*    msf   = (float*)   (ws + 860160);     //  96 KB
  float*    pms   = (float*)   (ws + 958464);     //   3 MB
  float*    pv1   = (float*)   (ws + 4104192);    //  48 MB

  k0_prep<<<64, 256, 0, stream>>>(Wq, Wv, keyp, wqh, wvh, keyh);
  k1_stats<<<dim3(NCHUNK, NSLAB), 256, 0, stream>>>(x, wqh, wvh, keyh, bq, bv, pv1, pms);
  k2_reduce<<<NSLAB, 256, 0, stream>>>(pv1, pms, msf, v1T);
  k3_out<<<dim3(NCHUNK, NSLAB), 256, 0, stream>>>(x, wqh, wvh, keyh, bq, bv,
                                                  alpha, beta, msf, v1T, out);
}
